// HGT_59622736003342
// MI455X (gfx1250) — matrix operand order repaired and checked by reading, not compiled, not run on hardware
//
#include <hip/hip_runtime.h>
#include <math.h>

#define N_NODES 50000
#define N_EDGES 200000
#define CH      256
#define NC      (N_NODES * CH)
#define RSQRT_D 0.0883883476483184f   // 1/sqrt(128)

typedef __attribute__((ext_vector_type(16))) __bf16 v16bf;
typedef __attribute__((ext_vector_type(8)))  __bf16 v8bf;
typedef __attribute__((ext_vector_type(8)))  float  v8f;

__device__ __forceinline__ unsigned short f2bf(float f) {
  unsigned u = __float_as_uint(f);
  unsigned r = ((u >> 16) & 1u) + 0x7FFFu;   // round-to-nearest-even
  return (unsigned short)((u + r) >> 16);
}

// ---------------- prep kernels ----------------

__global__ void cvt_bf16_kernel(const float* __restrict__ in,
                                unsigned short* __restrict__ out, int n) {
  int t = blockIdx.x * 256 + threadIdx.x;
  if (t < n) out[t] = f2bf(in[t]);
}

__global__ void zero_f32_kernel(float* __restrict__ p, int n) {
  int t = blockIdx.x * 256 + threadIdx.x;
  if (t < n) p[t] = 0.0f;
}

__global__ void init_seg_kernel(float* __restrict__ amax,
                                float* __restrict__ denom, int n) {
  int t = blockIdx.x * 256 + threadIdx.x;
  if (t < n) { amax[t] = -INFINITY; denom[t] = 0.0f; }
}

// Wout[k, h*128+e] = sum_d W[k, h*128+d] * rel[h, d, e]   (fuse a_rel/m_rel into weights)
// virtual row 256 computes the fused bias.
__global__ void combine_rel_kernel(const float* __restrict__ W,
                                   const float* __restrict__ b,
                                   const float* __restrict__ rel,
                                   float* __restrict__ Wout,
                                   float* __restrict__ bout) {
  int t = blockIdx.x * 256 + threadIdx.x;        // 0 .. 65791
  int col = t & 255;
  int row = t >> 8;                              // 0..256
  if (row > 256) return;
  int h = col >> 7;
  int e = col & 127;
  const float* relh = rel + h * (128 * 128) + e; // stride 128 over d
  if (row < 256) {
    const float* wr = W + row * 256 + h * 128;
    float s = 0.0f;
    #pragma unroll 8
    for (int d = 0; d < 128; ++d) s += wr[d] * relh[d * 128];
    Wout[row * 256 + col] = s;
  } else {
    const float* br = b + h * 128;
    float s = 0.0f;
    #pragma unroll 8
    for (int d = 0; d < 128; ++d) s += br[d] * relh[d * 128];
    bout[col] = s;
  }
}

// Pack a row-major fp32 [256(K),256(N)] matrix into bf16 WMMA B-fragments:
// P[((ntile*8 + ks)*32 + lane)*16 + i] = W[ks*32 + (lane>>4)*16 + i][ntile*16 + (lane&15)]
__global__ void pack_b_kernel(const float* __restrict__ W,
                              unsigned short* __restrict__ P) {
  int t = blockIdx.x * 256 + threadIdx.x;        // 65536
  int i    = t & 15;
  int lane = (t >> 4) & 31;
  int ks   = (t >> 9) & 7;
  int nt   = t >> 12;
  int k = ks * 32 + ((lane >> 4) << 4) + i;
  int n = (nt << 4) + (lane & 15);
  P[t] = f2bf(W[k * 256 + n]);
}

__global__ void gelu_bf16_kernel(const float* __restrict__ in,
                                 unsigned short* __restrict__ out, int n) {
  int t = blockIdx.x * 256 + threadIdx.x;
  if (t < n) {
    float x = in[t];
    float g = 0.5f * x * (1.0f + erff(x * 0.7071067811865476f));
    out[t] = f2bf(g);
  }
}

// ---------------- bf16 WMMA GEMM ----------------
// Y[nrows,256] = bf16(Xb) @ Bpacked + bias ; optionally gated residual:
//   Y = a*(..) + (1-a)*resid,  a = sigmoid(skip[0])
// One wave computes a 16(M) x 64(N) tile; K=256 in 8 steps of 32.
__global__ void __launch_bounds__(256)
gemm_bf16_kernel(const unsigned short* __restrict__ Xb,
                 const unsigned short* __restrict__ Bp,
                 const float* __restrict__ bias,
                 float* __restrict__ Y, int nrows, int gated,
                 const float* __restrict__ resid,
                 const float* __restrict__ skip) {
  int wave = (blockIdx.x * blockDim.x + threadIdx.x) >> 5;
  int lane = threadIdx.x & 31;
  int total = (nrows >> 4) * 4;
  if (wave >= total) return;
  int mt = wave >> 2;
  int ng = wave & 3;
  int row = (mt << 4) + (lane & 15);
  int kg  = lane >> 4;

  v8f acc0 = {}, acc1 = {}, acc2 = {}, acc3 = {};
  const unsigned short* xrow  = Xb + row * 256 + kg * 16;
  const unsigned short* bbase = Bp + ng * 16384 + lane * 16;

  #pragma unroll
  for (int ks = 0; ks < 8; ++ks) {
    const v8bf* pa = (const v8bf*)(const void*)(xrow + ks * 32);
    v8bf alo = pa[0];
    v8bf ahi = pa[1];   // +8 elements
    v16bf a = __builtin_shufflevector(alo, ahi, 0,1,2,3,4,5,6,7,8,9,10,11,12,13,14,15);
    v16bf b0 = *(const v16bf*)(const void*)(bbase + 0 * 4096 + ks * 512);
    v16bf b1 = *(const v16bf*)(const void*)(bbase + 1 * 4096 + ks * 512);
    v16bf b2 = *(const v16bf*)(const void*)(bbase + 2 * 4096 + ks * 512);
    v16bf b3 = *(const v16bf*)(const void*)(bbase + 3 * 4096 + ks * 512);
    acc0 = __builtin_amdgcn_wmma_f32_16x16x32_bf16(false, a, false, b0, (short)0, acc0, false, false);
    acc1 = __builtin_amdgcn_wmma_f32_16x16x32_bf16(false, a, false, b1, (short)0, acc1, false, false);
    acc2 = __builtin_amdgcn_wmma_f32_16x16x32_bf16(false, a, false, b2, (short)0, acc2, false, false);
    acc3 = __builtin_amdgcn_wmma_f32_16x16x32_bf16(false, a, false, b3, (short)0, acc3, false, false);
  }

  float a_gate = 1.0f, b_gate = 0.0f;
  if (gated) {
    float sv = skip[0];
    a_gate = 1.0f / (1.0f + expf(-sv));
    b_gate = 1.0f - a_gate;
  }
  v8f accs[4] = {acc0, acc1, acc2, acc3};
  #pragma unroll
  for (int j = 0; j < 4; ++j) {
    int ncol = ng * 64 + j * 16 + (lane & 15);
    float bs = bias[ncol];
    #pragma unroll
    for (int r = 0; r < 8; ++r) {
      int m = (mt << 4) + r + kg * 8;
      float v = accs[j][r] + bs;
      if (gated) v = a_gate * v + b_gate * resid[m * 256 + ncol];
      Y[m * 256 + ncol] = v;   // lanes 0-15 / 16-31 each store 16 contiguous floats
    }
  }
}

// ---------------- edge attention kernels ----------------

// one wave per edge: alpha[e,h] = (q[dst]·k_rel[src])_h * p_rel[h]/sqrt(D); atomic-max into amax
__global__ void __launch_bounds__(256)
edge_alpha_kernel(const int* __restrict__ eidx, const float* __restrict__ q,
                  const float* __restrict__ krel, const float* __restrict__ p_rel,
                  float* __restrict__ alpha, float* __restrict__ amax) {
  int wave = (blockIdx.x * blockDim.x + threadIdx.x) >> 5;
  int lane = threadIdx.x & 31;
  if (wave >= N_EDGES) return;
  int src = eidx[wave];
  int dst = eidx[N_EDGES + wave];
  int h = lane >> 4;
  int base = h * 128 + (lane & 15) * 8;
  const float4* qp = (const float4*)(const void*)(q    + (size_t)dst * 256 + base);
  const float4* kp = (const float4*)(const void*)(krel + (size_t)src * 256 + base);
  float4 q0 = qp[0], q1 = qp[1];
  float4 k0 = kp[0], k1 = kp[1];
  float s = q0.x*k0.x + q0.y*k0.y + q0.z*k0.z + q0.w*k0.w
          + q1.x*k1.x + q1.y*k1.y + q1.z*k1.z + q1.w*k1.w;
  s += __shfl_xor(s, 1);
  s += __shfl_xor(s, 2);
  s += __shfl_xor(s, 4);
  s += __shfl_xor(s, 8);
  if ((lane & 15) == 0) {
    float al = s * p_rel[h] * RSQRT_D;
    alpha[wave * 2 + h] = al;
    float* ap = amax + dst * 2 + h;
    if (al >= 0.0f) atomicMax((int*)ap, __float_as_int(al));
    else            atomicMin((unsigned int*)ap, __float_as_uint(al));
  }
}

__global__ void edge_exp_kernel(const int* __restrict__ eidx,
                                float* __restrict__ alpha,
                                const float* __restrict__ amax,
                                float* __restrict__ denom) {
  int t = blockIdx.x * 256 + threadIdx.x;
  if (t >= N_EDGES * 2) return;
  int e = t >> 1, h = t & 1;
  int dst = eidx[N_EDGES + e];
  float ex = expf(alpha[t] - amax[dst * 2 + h]);
  alpha[t] = ex;                       // overwrite alpha with exp()
  atomicAdd(denom + dst * 2 + h, ex);
}

// 64 threads per edge, float4 per thread: agg[dst] += v_rel[src] * w
__global__ void __launch_bounds__(256)
edge_scatter_kernel(const int* __restrict__ eidx, const float* __restrict__ alpha,
                    const float* __restrict__ denom, const float* __restrict__ vrel,
                    float* __restrict__ agg) {
  int t = blockIdx.x * blockDim.x + threadIdx.x;
  int e = t >> 6;
  if (e >= N_EDGES) return;
  int c4 = t & 63;
  int h = c4 >> 5;
  int src = eidx[e];
  int dst = eidx[N_EDGES + e];
  float w = alpha[e * 2 + h] / (denom[dst * 2 + h] + 1e-16f);
  float4 v = ((const float4*)(const void*)(vrel + (size_t)src * 256))[c4];
  float* ap = agg + (size_t)dst * 256 + c4 * 4;
  atomicAdd(ap + 0, v.x * w);
  atomicAdd(ap + 1, v.y * w);
  atomicAdd(ap + 2, v.z * w);
  atomicAdd(ap + 3, v.w * w);
}

// ---------------- host-side launch ----------------

extern "C" void kernel_launch(void* const* d_in, const int* in_sizes, int n_in,
                              void* d_out, int out_size, void* d_ws, size_t ws_size,
                              hipStream_t stream) {
  (void)in_sizes; (void)n_in; (void)out_size; (void)ws_size;

  const float* x[2]     = { (const float*)d_in[0], (const float*)d_in[1] };
  const int*   edges[3] = { (const int*)d_in[2], (const int*)d_in[3], (const int*)d_in[4] };

  const float *k_w[2], *k_b[2], *q_w[2], *q_b[2], *v_w[2], *v_b[2],
              *a_w[2], *a_b[2], *skip[2];
  for (int t = 0; t < 2; ++t) {
    int b = 5 + 9 * t;
    k_w[t] = (const float*)d_in[b + 0]; k_b[t] = (const float*)d_in[b + 1];
    q_w[t] = (const float*)d_in[b + 2]; q_b[t] = (const float*)d_in[b + 3];
    v_w[t] = (const float*)d_in[b + 4]; v_b[t] = (const float*)d_in[b + 5];
    a_w[t] = (const float*)d_in[b + 6]; a_b[t] = (const float*)d_in[b + 7];
    skip[t] = (const float*)d_in[b + 8];
  }
  const float *a_rel[3], *m_rel[3], *p_rel[3];
  for (int r = 0; r < 3; ++r) {
    int b = 23 + 3 * r;
    a_rel[r] = (const float*)d_in[b + 0];
    m_rel[r] = (const float*)d_in[b + 1];
    p_rel[r] = (const float*)d_in[b + 2];
  }
  const int rel_src[3] = {0, 0, 1};   // connects: h->h, has: h->v, affects: v->h
  const int rel_dst[3] = {0, 1, 0};

  // workspace layout
  char* ws = (char*)d_ws;
  size_t off = 0;
  auto alloc = [&](size_t bytes) -> void* {
    void* p = ws + off;
    off += (bytes + 255) & ~(size_t)255;
    return p;
  };
  unsigned short* XB[2];
  XB[0] = (unsigned short*)alloc((size_t)NC * 2);
  XB[1] = (unsigned short*)alloc((size_t)NC * 2);
  float* Q[2];
  Q[0] = (float*)alloc((size_t)NC * 4);
  Q[1] = (float*)alloc((size_t)NC * 4);
  float* KREL = (float*)alloc((size_t)NC * 4);
  float* VREL = (float*)alloc((size_t)NC * 4);
  float* AGG[2];
  AGG[0] = (float*)alloc((size_t)NC * 4);
  AGG[1] = (float*)alloc((size_t)NC * 4);
  float* ALPHA = (float*)alloc((size_t)N_EDGES * 2 * 4);
  float* AMAX  = (float*)alloc((size_t)N_NODES * 2 * 4);
  float* DENOM = (float*)alloc((size_t)N_NODES * 2 * 4);
  float* WCOMB = (float*)alloc((size_t)6 * 65536 * 4);
  float* BCOMB = (float*)alloc((size_t)6 * 256 * 4);
  unsigned short* BPACK = (unsigned short*)alloc((size_t)10 * 65536 * 2);

  dim3 blk(256);
  const int g_nc   = (NC + 255) / 256;
  const int g_gemm = ((N_NODES / 16) * 4 + 7) / 8;   // waves/8 per block
  const int g_seg  = (N_NODES * 2 + 255) / 256;

  // 0) convert activations to bf16
  cvt_bf16_kernel<<<g_nc, blk, 0, stream>>>(x[0], XB[0], NC);
  cvt_bf16_kernel<<<g_nc, blk, 0, stream>>>(x[1], XB[1], NC);

  // 1) fuse relation transforms into K/V weights, then pack all 10 B matrices
  for (int r = 0; r < 3; ++r) {
    int s = rel_src[r];
    combine_rel_kernel<<<257, blk, 0, stream>>>(k_w[s], k_b[s], a_rel[r],
                                                WCOMB + (size_t)r * 65536, BCOMB + r * 256);
    combine_rel_kernel<<<257, blk, 0, stream>>>(v_w[s], v_b[s], m_rel[r],
                                                WCOMB + (size_t)(3 + r) * 65536, BCOMB + (3 + r) * 256);
  }
  pack_b_kernel<<<256, blk, 0, stream>>>(q_w[0], BPACK + (size_t)0 * 65536);
  pack_b_kernel<<<256, blk, 0, stream>>>(q_w[1], BPACK + (size_t)1 * 65536);
  for (int r = 0; r < 3; ++r) {
    pack_b_kernel<<<256, blk, 0, stream>>>(WCOMB + (size_t)r * 65536,       BPACK + (size_t)(2 + r) * 65536);
    pack_b_kernel<<<256, blk, 0, stream>>>(WCOMB + (size_t)(3 + r) * 65536, BPACK + (size_t)(5 + r) * 65536);
  }
  pack_b_kernel<<<256, blk, 0, stream>>>(a_w[0], BPACK + (size_t)8 * 65536);
  pack_b_kernel<<<256, blk, 0, stream>>>(a_w[1], BPACK + (size_t)9 * 65536);

  // 2) zero aggregation buffers (workspace is poisoned)
  zero_f32_kernel<<<g_nc, blk, 0, stream>>>(AGG[0], NC);
  zero_f32_kernel<<<g_nc, blk, 0, stream>>>(AGG[1], NC);

  // 3) Q projections (WMMA)
  gemm_bf16_kernel<<<g_gemm, blk, 0, stream>>>(XB[0], BPACK + (size_t)0 * 65536, q_b[0],
                                               Q[0], N_NODES, 0, nullptr, nullptr);
  gemm_bf16_kernel<<<g_gemm, blk, 0, stream>>>(XB[1], BPACK + (size_t)1 * 65536, q_b[1],
                                               Q[1], N_NODES, 0, nullptr, nullptr);

  // 4) per-relation: fused K/V projections (WMMA) + edge segment-softmax + scatter
  for (int r = 0; r < 3; ++r) {
    int s = rel_src[r], d = rel_dst[r];
    init_seg_kernel<<<g_seg, blk, 0, stream>>>(AMAX, DENOM, N_NODES * 2);
    gemm_bf16_kernel<<<g_gemm, blk, 0, stream>>>(XB[s], BPACK + (size_t)(2 + r) * 65536,
                                                 BCOMB + r * 256, KREL, N_NODES, 0, nullptr, nullptr);
    gemm_bf16_kernel<<<g_gemm, blk, 0, stream>>>(XB[s], BPACK + (size_t)(5 + r) * 65536,
                                                 BCOMB + (3 + r) * 256, VREL, N_NODES, 0, nullptr, nullptr);
    edge_alpha_kernel<<<N_EDGES / 8, blk, 0, stream>>>(edges[r], Q[d], KREL, p_rel[r], ALPHA, AMAX);
    edge_exp_kernel<<<(N_EDGES * 2 + 255) / 256, blk, 0, stream>>>(edges[r], ALPHA, AMAX, DENOM);
    edge_scatter_kernel<<<N_EDGES / 4, blk, 0, stream>>>(edges[r], ALPHA, DENOM, VREL, AGG[d]);
  }

  // 5) GELU -> bf16, output projection (WMMA) with gated residual, write d_out
  float* outp = (float*)d_out;
  for (int t = 0; t < 2; ++t) {
    unsigned short* GB = (t == 0) ? (unsigned short*)KREL : (unsigned short*)VREL; // reuse scratch
    gelu_bf16_kernel<<<g_nc, blk, 0, stream>>>(AGG[t], GB, NC);
    gemm_bf16_kernel<<<g_gemm, blk, 0, stream>>>(GB, BPACK + (size_t)(8 + t) * 65536, a_b[t],
                                                 outp + (size_t)t * NC, N_NODES, 1, x[t], skip[t]);
  }
}
